// ASLREModel_5540507811927
// MI455X (gfx1250) — matrix operand order repaired and checked by reading, not compiled, not run on hardware
//
#include <hip/hip_runtime.h>
#include <hip/hip_bf16.h>
#include <math.h>

// ---------------------------------------------------------------------------
// Model dims
// ---------------------------------------------------------------------------
#define BB     128
#define NN     256
#define DEMB   256
#define NODE   512
#define LL     2
#define TOPK   8
#define HID    512
#define LABELS 19

typedef __attribute__((ext_vector_type(16))) __bf16 v16bf;
typedef __attribute__((ext_vector_type(8)))  __bf16 v8bf;
typedef __attribute__((ext_vector_type(8)))  float  v8f;

// ---------------------------------------------------------------------------
// K1: embedding gather (directly into x[:, :, 0:256]) + masked means -> entcat
// grid(B), block(256)  -- thread = d
// ---------------------------------------------------------------------------
__global__ __launch_bounds__(256) void embed_kernel(
    const int* __restrict__ ids, const float* __restrict__ e1,
    const float* __restrict__ e2, const float* __restrict__ emb,
    float* __restrict__ x, float* __restrict__ entcat) {
  int b = blockIdx.x, d = threadIdx.x;
  float a1 = 0.f, a2 = 0.f, c1 = 0.f, c2 = 0.f;
  for (int n = 0; n < NN; ++n) {
    int id = ids[b * NN + n];
    float v = emb[(size_t)id * DEMB + d];
    x[((size_t)b * NN + n) * NODE + d] = v;
    float m1 = e1[b * NN + n], m2 = e2[b * NN + n];
    a1 += m1 * v; a2 += m2 * v; c1 += m1; c2 += m2;
  }
  entcat[(size_t)b * 512 + d]       = a1 / (c1 + 1e-13f);
  entcat[(size_t)b * 512 + 256 + d] = a2 / (c2 + 1e-13f);
}

// ---------------------------------------------------------------------------
// K2: ent = entcat[B,512] @ ep_w[512,256] + ep_b   (thread per (b,j))
// ---------------------------------------------------------------------------
__global__ __launch_bounds__(256) void entproj_kernel(
    const float* __restrict__ entcat, const float* __restrict__ w,
    const float* __restrict__ bias, float* __restrict__ ent) {
  int idx = blockIdx.x * 256 + threadIdx.x;
  int b = idx >> 8, j = idx & 255;
  const float* er = entcat + (size_t)b * 512;
  float acc = bias[j];
  for (int d = 0; d < 512; ++d) acc += er[d] * w[d * 256 + j];
  ent[idx] = acc;
}

// K3: x[:, :, 256:512] = ent broadcast over n.  grid(N, B), block(256)
__global__ __launch_bounds__(256) void bcast_kernel(
    const float* __restrict__ ent, float* __restrict__ x) {
  int n = blockIdx.x, b = blockIdx.y, j = threadIdx.x;
  x[((size_t)b * NN + n) * NODE + DEMB + j] = ent[b * DEMB + j];
}

// ---------------------------------------------------------------------------
// K4: per-layer ternary quantization, stored transposed bf16:
// qwT[j,k] = quant(w[k,j]).  grid(512), block(512)
// ---------------------------------------------------------------------------
__global__ __launch_bounds__(512) void quant_kernel(
    const float* __restrict__ w, __bf16* __restrict__ qwT) {
  int k = blockIdx.x, j = threadIdx.x;
  float v = w[k * NODE + j];
  float q = (v > 0.1f) ? 0.1f : ((v < -0.1f) ? -0.1f : 0.0f);
  qwT[(size_t)j * NODE + k] = (__bf16)q;
}

// ---------------------------------------------------------------------------
// K5: row norms -> xbf (bf16 x), xnbf (bf16 x/||x||).  grid(N,B), block(256)
// ---------------------------------------------------------------------------
__global__ __launch_bounds__(256) void prep_kernel(
    const float* __restrict__ x, __bf16* __restrict__ xbf,
    __bf16* __restrict__ xnbf) {
  __shared__ float red[256];
  int n = blockIdx.x, b = blockIdx.y, t = threadIdx.x;
  size_t off = ((size_t)b * NN + n) * NODE;
  float v0 = x[off + t], v1 = x[off + t + 256];
  red[t] = v0 * v0 + v1 * v1;
  __syncthreads();
  for (int s = 128; s > 0; s >>= 1) {
    if (t < s) red[t] += red[t + s];
    __syncthreads();
  }
  float nrm = sqrtf(red[0]);
  nrm = nrm > 1e-12f ? nrm : 1e-12f;
  float inv = 1.f / nrm;
  xbf[off + t]        = (__bf16)v0;
  xbf[off + t + 256]  = (__bf16)v1;
  xnbf[off + t]       = (__bf16)(v0 * inv);
  xnbf[off + t + 256] = (__bf16)(v1 * inv);
}

// ---------------------------------------------------------------------------
// K6: batched NT-GEMM with V_WMMA_F32_16X16X32_BF16, register-blocked.
// C[M,N] = A[M,K] * BT[N,K]^T.
// Each wave owns one 16-row M tile and FOUR 16-col N tiles (C = 16x64):
// the A fragment is loaded once per k-step and reused for 4 WMMAs, cutting
// A-stream traffic 4x (the big win for the 32768x512x512 x@qw GEMM).
// 4 waves per block cover 4 consecutive M tiles; B fragments are identical
// across waves and served by L0/L2 (BT is L2-resident: qwT = 512 KB).
//
// A lane layout (16-bit A 16x32): lane m=l&15, h=l>>4, elem e:
//   k = kk + h*16 + e                  -> one contiguous 16-elem chunk
// B lane layout (16-bit B 32x16): lane n=l&15, elem e: k = kk + h*16 + e
// C layout: VGPR r -> row r + 8*h, col = lane&15.
// ---------------------------------------------------------------------------
__global__ __launch_bounds__(128) void gemm_nt_bf16(
    const __bf16* __restrict__ A, const __bf16* __restrict__ BT,
    float* __restrict__ C, int M, int N, int K,
    long long sA, long long sB, long long sC) {
  int lane = threadIdx.x & 31;
  int wave = threadIdx.x >> 5;
  int tm  = blockIdx.x * 4 + wave;   // wave's 16-row M tile
  int tn0 = blockIdx.y * 4;          // block's first 16-col N tile (covers 64)
  int bz  = blockIdx.z;
  if (tm * 16 >= M) return;
  int m = lane & 15, h = lane >> 4;

  const __bf16* arow = A + (size_t)bz * sA + ((size_t)tm * 16 + m) * K;
  const __bf16* br0  = BT + (size_t)bz * sB + ((size_t)(tn0 + 0) * 16 + m) * K;
  const __bf16* br1  = BT + (size_t)bz * sB + ((size_t)(tn0 + 1) * 16 + m) * K;
  const __bf16* br2  = BT + (size_t)bz * sB + ((size_t)(tn0 + 2) * 16 + m) * K;
  const __bf16* br3  = BT + (size_t)bz * sB + ((size_t)(tn0 + 3) * 16 + m) * K;

  v8f c0 = {0.f,0.f,0.f,0.f,0.f,0.f,0.f,0.f};
  v8f c1 = c0, c2 = c0, c3 = c0;

  #pragma unroll 2
  for (int kk = 0; kk < K; kk += 32) {
    // prefetch next A chunk into cache (global_prefetch_b8)
    __builtin_prefetch(arow + kk + 128, 0, 1);

    v8bf a0 = *(const v8bf*)(arow + kk + h * 16);
    v8bf a1 = *(const v8bf*)(arow + kk + h * 16 + 8);
    v16bf av;
    #pragma unroll
    for (int e = 0; e < 8; ++e) { av[e] = a0[e]; av[e + 8] = a1[e]; }

    v8bf t0, t1;
    v16bf bv;
    t0 = *(const v8bf*)(br0 + kk + h * 16);
    t1 = *(const v8bf*)(br0 + kk + h * 16 + 8);
    #pragma unroll
    for (int e = 0; e < 8; ++e) { bv[e] = t0[e]; bv[e + 8] = t1[e]; }
    c0 = __builtin_amdgcn_wmma_f32_16x16x32_bf16(false, av, false, bv, (short)0, c0, false, false);

    t0 = *(const v8bf*)(br1 + kk + h * 16);
    t1 = *(const v8bf*)(br1 + kk + h * 16 + 8);
    #pragma unroll
    for (int e = 0; e < 8; ++e) { bv[e] = t0[e]; bv[e + 8] = t1[e]; }
    c1 = __builtin_amdgcn_wmma_f32_16x16x32_bf16(false, av, false, bv, (short)0, c1, false, false);

    t0 = *(const v8bf*)(br2 + kk + h * 16);
    t1 = *(const v8bf*)(br2 + kk + h * 16 + 8);
    #pragma unroll
    for (int e = 0; e < 8; ++e) { bv[e] = t0[e]; bv[e + 8] = t1[e]; }
    c2 = __builtin_amdgcn_wmma_f32_16x16x32_bf16(false, av, false, bv, (short)0, c2, false, false);

    t0 = *(const v8bf*)(br3 + kk + h * 16);
    t1 = *(const v8bf*)(br3 + kk + h * 16 + 8);
    #pragma unroll
    for (int e = 0; e < 8; ++e) { bv[e] = t0[e]; bv[e + 8] = t1[e]; }
    c3 = __builtin_amdgcn_wmma_f32_16x16x32_bf16(false, av, false, bv, (short)0, c3, false, false);
  }

  float* crow = C + (size_t)bz * sC + ((size_t)tm * 16 + h * 8) * N + m;
  #pragma unroll
  for (int r = 0; r < 8; ++r) {
    crow[(size_t)r * N + (tn0 + 0) * 16] = c0[r];
    crow[(size_t)r * N + (tn0 + 1) * 16] = c1[r];
    crow[(size_t)r * N + (tn0 + 2) * 16] = c2[r];
    crow[(size_t)r * N + (tn0 + 3) * 16] = c3[r];
  }
}

// ---------------------------------------------------------------------------
// K7: top-k(8) per sim row -> scatter into dense mask row (rest zero).
// JAX tie-break: lower index wins on equal values. grid(N,B), block(256)
// ---------------------------------------------------------------------------
__global__ __launch_bounds__(256) void topk_kernel(
    const float* __restrict__ sim, float* __restrict__ adjm) {
  __shared__ float sval[256];
  __shared__ float sv[256];
  __shared__ int   si[256];
  int n = blockIdx.x, b = blockIdx.y, t = threadIdx.x;
  const float* srow = sim + ((size_t)b * NN + n) * NN;
  float* arow = adjm + ((size_t)b * NN + n) * NN;
  arow[t] = 0.f;
  sval[t] = srow[t];
  __syncthreads();
  for (int kk = 0; kk < TOPK; ++kk) {
    sv[t] = sval[t]; si[t] = t;
    __syncthreads();
    for (int s = 128; s > 0; s >>= 1) {
      if (t < s) {
        float v2 = sv[t + s]; int i2 = si[t + s];
        if (v2 > sv[t] || (v2 == sv[t] && i2 < si[t])) { sv[t] = v2; si[t] = i2; }
      }
      __syncthreads();
    }
    if (t == 0) { arow[si[0]] = sv[0]; sval[si[0]] = -INFINITY; }
    __syncthreads();
  }
}

// ---------------------------------------------------------------------------
// K8: fused conv1(1->32,3x3) + BN(eval) + ReLU + conv2(32->1,3x3), LDS-staged,
// so h[B,32,256,512] (2.1 GB) never reaches HBM.
// Tile 4x64 outputs, halo'd x tile 8x68, h tile 6x66x32 in LDS (pad stride 33).
// grid(512/64, 256/4, B), block(256)
// ---------------------------------------------------------------------------
#define CTH 4
#define CTW 64
#define CXH 8
#define CXW 68
#define CHH 6
#define CHW 66
#define SHP 33

__global__ __launch_bounds__(256) void conv_kernel(
    const float* __restrict__ x, const float* __restrict__ c1w,
    const float* __restrict__ c1b, const float* __restrict__ bng,
    const float* __restrict__ bnb, const float* __restrict__ c2w,
    const float* __restrict__ c2b, float* __restrict__ out) {
  __shared__ float sx[CXH * CXW];
  __shared__ float sh[CHH * CHW * SHP];
  __shared__ float sc1[32 * 9], sc2[32 * 9], sbs[32], sbb[32], sc1bias[32];
  int t = threadIdx.x;
  int tx = blockIdx.x, ty = blockIdx.y, b = blockIdx.z;
  int y0 = ty * CTH, x0 = tx * CTW;

  for (int i = t; i < 288; i += 256) { sc1[i] = c1w[i]; sc2[i] = c2w[i]; }
  if (t < 32) {
    sbs[t] = bng[t] * rsqrtf(1.0f + 1e-5f);
    sbb[t] = bnb[t];
    sc1bias[t] = c1b[t];
  }
  for (int i = t; i < CXH * CXW; i += 256) {
    int ly = i / CXW, lx = i % CXW;
    int gy = y0 - 2 + ly, gx = x0 - 2 + lx;
    float v = 0.f;
    if (gy >= 0 && gy < NN && gx >= 0 && gx < NODE)
      v = x[((size_t)b * NN + gy) * NODE + gx];
    sx[i] = v;
  }
  __syncthreads();

  // phase 1: h = relu(bn(conv1(x)))  over 6x66 halo'd region, 32 channels.
  // h outside the true image is ZERO (conv2 same-padding semantics).
  for (int i = t; i < CHH * CHW * 32; i += 256) {
    int pos = i >> 5, oc = i & 31;
    int hy = pos / CHW, hx = pos % CHW;
    int gy = y0 - 1 + hy, gx = x0 - 1 + hx;
    float val = 0.f;
    if (gy >= 0 && gy < NN && gx >= 0 && gx < NODE) {
      float acc = sc1bias[oc];
      const float* w = &sc1[oc * 9];
      #pragma unroll
      for (int dy = 0; dy < 3; ++dy)
        #pragma unroll
        for (int dx = 0; dx < 3; ++dx)
          acc += w[dy * 3 + dx] * sx[(hy + dy) * CXW + (hx + dx)];
      acc = acc * sbs[oc] + sbb[oc];
      val = acc > 0.f ? acc : 0.f;
    }
    sh[pos * SHP + oc] = val;
  }
  __syncthreads();

  // phase 2: conv2 32->1
  float bias2 = c2b[0];
  for (int o = t; o < CTH * CTW; o += 256) {
    int oy = o / CTW, ox = o % CTW;
    float acc = bias2;
    #pragma unroll
    for (int dy = 0; dy < 3; ++dy)
      #pragma unroll
      for (int dx = 0; dx < 3; ++dx) {
        const float* hp = &sh[((oy + dy) * CHW + (ox + dx)) * SHP];
        #pragma unroll
        for (int oc = 0; oc < 32; ++oc)
          acc += sc2[oc * 9 + dy * 3 + dx] * hp[oc];
      }
    out[((size_t)b * NN + y0 + oy) * NODE + x0 + ox] = acc;
  }
}

// ---------------------------------------------------------------------------
// K9: fused sparse adjacency aggregation + ReLU + conv add + gate + residual.
// adj row = (top-k row + top-k column)/2, zeros skipped; deterministic loop order.
// grid(N,B), block(256) -- thread handles d and d+256.
// ---------------------------------------------------------------------------
__global__ __launch_bounds__(256) void combine_kernel(
    const float* __restrict__ xw, const float* __restrict__ adjm,
    const float* __restrict__ convout, const float* __restrict__ gcn_b,
    const float* __restrict__ gate_w, const float* __restrict__ gate_b,
    float* __restrict__ x) {
  __shared__ float a_s[256];
  __shared__ float red[256];
  int n = blockIdx.x, b = blockIdx.y, t = threadIdx.x;
  float an = adjm[((size_t)b * NN + n) * NN + t];
  float am = adjm[((size_t)b * NN + t) * NN + n];
  a_s[t] = 0.5f * (an + am);
  __syncthreads();

  int d0 = t, d1 = t + 256;
  size_t rowoff = ((size_t)b * NN + n) * NODE;
  float acc0 = 0.f, acc1 = 0.f;
  for (int m = 0; m < NN; ++m) {
    float coef = a_s[m];
    if (coef != 0.f) {   // uniform across block -> cheap scalar branch
      const float* xwr = xw + ((size_t)b * NN + m) * NODE;
      acc0 += coef * xwr[d0];
      acc1 += coef * xwr[d1];
    }
  }
  float g0 = acc0 + gcn_b[d0]; g0 = g0 > 0.f ? g0 : 0.f;
  float g1 = acc1 + gcn_b[d1]; g1 = g1 > 0.f ? g1 : 0.f;
  float comb0 = g0 + convout[rowoff + d0];
  float comb1 = g1 + convout[rowoff + d1];
  float r0 = x[rowoff + d0], r1 = x[rowoff + d1];

  red[t] = r0 * gate_w[d0] + r1 * gate_w[d1] +
           comb0 * gate_w[NODE + d0] + comb1 * gate_w[NODE + d1];
  __syncthreads();
  for (int s = 128; s > 0; s >>= 1) {
    if (t < s) red[t] += red[t + s];
    __syncthreads();
  }
  float g = 1.f / (1.f + expf(-(red[0] + gate_b[0])));
  x[rowoff + d0] = g * r0 + (1.f - g) * comb0;
  x[rowoff + d1] = g * r1 + (1.f - g) * comb1;
}

// ---------------------------------------------------------------------------
// K10-12: pool + classifier (tiny; exact f32)
// ---------------------------------------------------------------------------
__global__ __launch_bounds__(256) void pool_kernel(
    const float* __restrict__ x, float* __restrict__ pooled) {
  int b = blockIdx.x, t = threadIdx.x;
  float a0 = 0.f, a1 = 0.f;
  for (int n = 0; n < NN; ++n) {
    size_t off = ((size_t)b * NN + n) * NODE;
    a0 += x[off + t];
    a1 += x[off + t + 256];
  }
  pooled[b * NODE + t] = a0 * (1.f / NN);
  pooled[b * NODE + t + 256] = a1 * (1.f / NN);
}

__global__ __launch_bounds__(256) void fc1_kernel(
    const float* __restrict__ pooled, const float* __restrict__ w,
    const float* __restrict__ bias, float* __restrict__ hid) {
  int idx = blockIdx.x * 256 + threadIdx.x;  // B*HID
  int b = idx >> 9, j = idx & 511;
  const float* pr = pooled + (size_t)b * NODE;
  float acc = bias[j];
  for (int d = 0; d < NODE; ++d) acc += pr[d] * w[d * HID + j];
  hid[idx] = acc > 0.f ? acc : 0.f;
}

__global__ __launch_bounds__(256) void fc2_kernel(
    const float* __restrict__ hid, const float* __restrict__ w,
    const float* __restrict__ bias, float* __restrict__ out) {
  int idx = blockIdx.x * 256 + threadIdx.x;
  if (idx >= BB * LABELS) return;
  int b = idx / LABELS, l = idx % LABELS;
  const float* hr = hid + (size_t)b * HID;
  float acc = bias[l];
  for (int j = 0; j < HID; ++j) acc += hr[j] * w[j * LABELS + l];
  out[idx] = acc;
}

// ---------------------------------------------------------------------------
// Host orchestration
// ---------------------------------------------------------------------------
extern "C" void kernel_launch(void* const* d_in, const int* in_sizes, int n_in,
                              void* d_out, int out_size, void* d_ws, size_t ws_size,
                              hipStream_t stream) {
  const int*   ids    = (const int*)d_in[0];
  const float* e1     = (const float*)d_in[1];
  const float* e2     = (const float*)d_in[2];
  const float* emb    = (const float*)d_in[3];
  const float* ep_w   = (const float*)d_in[4];
  const float* ep_b   = (const float*)d_in[5];
  const float* gcn_w  = (const float*)d_in[6];
  const float* gcn_b  = (const float*)d_in[7];
  const float* c1_w   = (const float*)d_in[8];
  const float* c1_b   = (const float*)d_in[9];
  const float* bn_g   = (const float*)d_in[10];
  const float* bn_b   = (const float*)d_in[11];
  const float* c2_w   = (const float*)d_in[12];
  const float* c2_b   = (const float*)d_in[13];
  const float* gate_w = (const float*)d_in[14];
  const float* gate_b = (const float*)d_in[15];
  const float* cl_w1  = (const float*)d_in[16];
  const float* cl_b1  = (const float*)d_in[17];
  const float* cl_w2  = (const float*)d_in[18];
  const float* cl_b2  = (const float*)d_in[19];

  char* p = (char*)d_ws;
  auto carve = [&](size_t bytes) -> void* {
    void* r = (void*)p;
    p += (bytes + 255) & ~(size_t)255;
    return r;
  };
  const size_t XE = (size_t)BB * NN * NODE;        // 16.7M elems
  const size_t SE = (size_t)BB * NN * NN;          // 8.4M elems
  float*  x       = (float*)carve(XE * 4);
  float*  xw      = (float*)carve(XE * 4);
  float*  convout = (float*)carve(XE * 4);
  float*  sim     = (float*)carve(SE * 4);
  float*  adjm    = (float*)carve(SE * 4);
  __bf16* xbf     = (__bf16*)carve(XE * 2);
  __bf16* xnbf    = (__bf16*)carve(XE * 2);
  __bf16* qwT     = (__bf16*)carve((size_t)NODE * NODE * 2);
  float*  entcat  = (float*)carve((size_t)BB * 512 * 4);
  float*  ent     = (float*)carve((size_t)BB * DEMB * 4);
  float*  pooled  = (float*)carve((size_t)BB * NODE * 4);
  float*  hidden  = (float*)carve((size_t)BB * HID * 4);

  embed_kernel<<<BB, 256, 0, stream>>>(ids, e1, e2, emb, x, entcat);
  entproj_kernel<<<(BB * DEMB) / 256, 256, 0, stream>>>(entcat, ep_w, ep_b, ent);
  bcast_kernel<<<dim3(NN, BB), 256, 0, stream>>>(ent, x);

  for (int i = 0; i < LL; ++i) {
    quant_kernel<<<NODE, NODE, 0, stream>>>(gcn_w + (size_t)i * NODE * NODE, qwT);
    prep_kernel<<<dim3(NN, BB), 256, 0, stream>>>(x, xbf, xnbf);
    // sim = xn . xn^T   (batched, M=N=256, K=512); wave computes 16x64 of C
    gemm_nt_bf16<<<dim3((NN / 16) / 4, NN / 64, BB), 128, 0, stream>>>(
        xnbf, xnbf, sim, NN, NN, NODE,
        (long long)NN * NODE, (long long)NN * NODE, (long long)NN * NN);
    topk_kernel<<<dim3(NN, BB), 256, 0, stream>>>(sim, adjm);
    // xw = x @ qw   (M = B*N = 32768, N = K = 512)
    gemm_nt_bf16<<<dim3(((BB * NN) / 16) / 4, NODE / 64, 1), 128, 0, stream>>>(
        xbf, qwT, xw, BB * NN, NODE, NODE, 0, 0, 0);
    conv_kernel<<<dim3(NODE / CTW, NN / CTH, BB), 256, 0, stream>>>(
        x, c1_w + (size_t)i * 288, c1_b + (size_t)i * 32,
        bn_g + (size_t)i * 32, bn_b + (size_t)i * 32,
        c2_w + (size_t)i * 288, c2_b + (size_t)i, convout);
    combine_kernel<<<dim3(NN, BB), 256, 0, stream>>>(
        xw, adjm, convout, gcn_b + (size_t)i * NODE, gate_w, gate_b, x);
  }

  pool_kernel<<<BB, 256, 0, stream>>>(x, pooled);
  fc1_kernel<<<(BB * HID) / 256, 256, 0, stream>>>(pooled, cl_w1, cl_b1, hidden);
  fc2_kernel<<<(BB * LABELS + 255) / 256, 256, 0, stream>>>(hidden, cl_w2, cl_b2,
                                                            (float*)d_out);
}
